// RecurrentBlock_3049426780600
// MI455X (gfx1250) — matrix operand order repaired and checked by reading, not compiled, not run on hardware
//
#include <hip/hip_runtime.h>
#include <stdint.h>
#include <stddef.h>

// Problem constants (match reference)
#define BT   8192      // B*T
#define Bn   4
#define Tn   2048
#define Dm   1024
#define HIDm 2048
#define EPSf 1e-6f

typedef __attribute__((ext_vector_type(16))) __bf16 v16bf;
typedef __attribute__((ext_vector_type(8)))  float  v8f;

__device__ __forceinline__ uint16_t f32_to_bf16(float f) {
  union { float f; uint32_t u; } c; c.f = f;
  uint32_t u = c.u;
  return (uint16_t)((u + 0x7FFFu + ((u >> 16) & 1u)) >> 16);
}

// Async global->LDS 16B copy (CDNA5 async engine, tracked by ASYNCcnt).
__device__ __forceinline__ void async_g2l_b128(unsigned lds_off,
                                               const uint16_t* gaddr) {
  asm volatile("global_load_async_to_lds_b128 %0, %1, off"
               :: "v"(lds_off), "v"(gaddr) : "memory");
}
__device__ __forceinline__ void wait_asynccnt0() {
  asm volatile("s_wait_asynccnt 0x0" ::: "memory");
}

// ---------------------------------------------------------------------------
// Pack an f32 row-major KxN weight into the WMMA B-fragment layout for
// v_wmma_f32_16x16x32_bf16. Fragment (kt,nt): lane L holds 16 bf16 with
// n = nt*16 + (L&15), K = kt*32 + (L>>4)*16 + e. Lane data contiguous (32B),
// lanes contiguous -> one fragment = 1KB contiguous block.
// ---------------------------------------------------------------------------
__global__ void pack_b_kernel(const float* __restrict__ W,
                              uint16_t* __restrict__ P, int K, int N) {
  int tid = blockIdx.x * blockDim.x + threadIdx.x;
  int KT = K >> 5;
  int total = KT * (N >> 4) * 32;
  if (tid >= total) return;
  int L    = tid & 31;
  int tile = tid >> 5;
  int kt   = tile % KT;
  int nt   = tile / KT;
  int n     = (nt << 4) + (L & 15);
  int kbase = (kt << 5) + ((L >> 4) << 4);
  uint16_t* outp = P + ((size_t)tid << 4);
#pragma unroll
  for (int e = 0; e < 16; ++e)
    outp[e] = f32_to_bf16(W[(size_t)(kbase + e) * N + n]);
}

// ---------------------------------------------------------------------------
// RMSNorm (one block per row). Writes bf16 h; optionally fuses the beta
// matvec: beta = sigmoid(h . beta_w + beta_b).
// ---------------------------------------------------------------------------
__global__ __launch_bounds__(256)
void rmsnorm_kernel(const float* __restrict__ x, const float* __restrict__ w,
                    uint16_t* __restrict__ hbf,
                    const float* __restrict__ beta_w,
                    const float* __restrict__ beta_b,
                    float* __restrict__ beta_out) {
  __shared__ float red[256];
  int row = blockIdx.x;
  const float* xr = x + (size_t)row * Dm;
  float xs[4];
  float ss = 0.f;
#pragma unroll
  for (int i = 0; i < 4; ++i) {
    xs[i] = xr[threadIdx.x + i * 256];
    ss += xs[i] * xs[i];
  }
  red[threadIdx.x] = ss;
  __syncthreads();
  for (int s = 128; s > 0; s >>= 1) {
    if (threadIdx.x < s) red[threadIdx.x] += red[threadIdx.x + s];
    __syncthreads();
  }
  float rs = rsqrtf(red[0] * (1.0f / (float)Dm) + EPSf);
  __syncthreads();
  float dot = 0.f;
#pragma unroll
  for (int i = 0; i < 4; ++i) {
    int d = threadIdx.x + i * 256;
    float h = xs[i] * rs * w[d];
    hbf[(size_t)row * Dm + d] = f32_to_bf16(h);
    if (beta_w) dot += h * beta_w[d];
  }
  if (beta_w) {
    red[threadIdx.x] = dot;
    __syncthreads();
    for (int s = 128; s > 0; s >>= 1) {
      if (threadIdx.x < s) red[threadIdx.x] += red[threadIdx.x + s];
      __syncthreads();
    }
    if (threadIdx.x == 0) {
      float z = red[0] + beta_b[0];
      beta_out[row] = 1.f / (1.f + __expf(-z));
    }
  }
}

// ---------------------------------------------------------------------------
// bf16 WMMA GEMM: C(f32, MxN) (+)= A(bf16 row-major MxK) @ Bpacked(KxN).
// Block = 256 threads = 8 waves (4x2); wave computes 32x64 (2x4 WMMA tiles),
// block computes 128x128. A tile (128x32 bf16) is staged in LDS with
// double-buffered async global->LDS b128 copies; rows padded to 80B so the
// b128 fragment reads hit 16 distinct DS banks. B fragments are direct,
// fully coalesced 1KB packed blocks.
// ---------------------------------------------------------------------------
#define AROW 40   // bf16 elements per padded LDS row (80B)

__global__ __launch_bounds__(256)
void gemm_bf16_kernel(const uint16_t* __restrict__ A,
                      const uint16_t* __restrict__ Bp,
                      float* __restrict__ C,
                      int M, int N, int K, int accumulate) {
  __shared__ uint16_t Abuf[2][128 * AROW];

  int tid  = threadIdx.x;
  int lane = tid & 31;
  int wid  = tid >> 5;
  int wm = wid >> 1, wn = wid & 1;
  int mBlock = blockIdx.y * 128;
  int mBase  = mBlock + wm * 32;
  int nBase  = blockIdx.x * 128 + wn * 64;
  int half = lane >> 4, l15 = lane & 15;
  int KT = K >> 5;

  // Async-copy assignment: 512 chunks of 16B (128 rows x 4 segs), 2/thread.
  int s0 = tid & 3;
  int r0 = tid >> 2;
  int r1 = 64 + r0;
  const uint16_t* gA0 = A + (size_t)(mBlock + r0) * K + s0 * 8;
  const uint16_t* gA1 = A + (size_t)(mBlock + r1) * K + s0 * 8;
  unsigned lA0 = (unsigned)(size_t)(void*)&Abuf[0][r0 * AROW + s0 * 8];
  unsigned lA1 = (unsigned)(size_t)(void*)&Abuf[0][r1 * AROW + s0 * 8];
  const unsigned bufStride = (unsigned)(128 * AROW * sizeof(uint16_t));

  v8f acc[2][4];
#pragma unroll
  for (int a = 0; a < 2; ++a)
#pragma unroll
    for (int b = 0; b < 4; ++b)
#pragma unroll
      for (int r = 0; r < 8; ++r) acc[a][b][r] = 0.f;

  const uint16_t* bptr[4];
#pragma unroll
  for (int nb = 0; nb < 4; ++nb)
    bptr[nb] = Bp + ((size_t)(((nBase >> 4) + nb) * KT) * 32 + lane) * 16;

  // Prologue: stage kt=0 into buffer 0.
  async_g2l_b128(lA0, gA0);
  async_g2l_b128(lA1, gA1);

  for (int kt = 0; kt < KT; ++kt) {
    wait_asynccnt0();
    __syncthreads();
    if (kt + 1 < KT) {
      unsigned lb = ((kt + 1) & 1) ? bufStride : 0u;
      async_g2l_b128(lA0 + lb, gA0 + (size_t)(kt + 1) * 32);
      async_g2l_b128(lA1 + lb, gA1 + (size_t)(kt + 1) * 32);
#pragma unroll
      for (int nb = 0; nb < 4; ++nb)
        __builtin_prefetch(bptr[nb] + (size_t)(kt + 1) * 512, 0, 1);
    }

    union Frag { v16bf v; uint4 u[2]; };
    Frag bfrag[4];
#pragma unroll
    for (int nb = 0; nb < 4; ++nb) {
      const uint4* bp = (const uint4*)(bptr[nb] + (size_t)kt * 512);
      bfrag[nb].u[0] = bp[0];
      bfrag[nb].u[1] = bp[1];
    }
    Frag af[2];
    const uint16_t* ab = &Abuf[kt & 1][0];
    const uint16_t* ar0 = ab + (wm * 32 + l15) * AROW + half * 16;
    const uint16_t* ar1 = ab + (wm * 32 + 16 + l15) * AROW + half * 16;
    af[0].u[0] = *(const uint4*)ar0;
    af[0].u[1] = *(const uint4*)(ar0 + 8);
    af[1].u[0] = *(const uint4*)ar1;
    af[1].u[1] = *(const uint4*)(ar1 + 8);

#pragma unroll
    for (int a = 0; a < 2; ++a)
#pragma unroll
      for (int b = 0; b < 4; ++b)
        acc[a][b] = __builtin_amdgcn_wmma_f32_16x16x32_bf16(
            false, af[a].v, false, bfrag[b].v, (short)0, acc[a][b],
            false, false);
  }

#pragma unroll
  for (int a = 0; a < 2; ++a)
#pragma unroll
    for (int b = 0; b < 4; ++b) {
      int col = nBase + b * 16 + l15;
#pragma unroll
      for (int r = 0; r < 8; ++r) {
        int rowi = mBase + a * 16 + r + half * 8;
        size_t idx = (size_t)rowi * N + col;
        float vv = acc[a][b][r];
        if (accumulate) C[idx] += vv; else C[idx] = vv;
      }
    }
  (void)M;
}

// ---------------------------------------------------------------------------
// k <- k / max(||k||, 1e-12), one block per row (in workspace, not input)
// ---------------------------------------------------------------------------
__global__ __launch_bounds__(256)
void knorm_kernel(float* __restrict__ kk) {
  __shared__ float red[256];
  int row = blockIdx.x;
  float* kr = kk + (size_t)row * Dm;
  float vs[4];
  float ss = 0.f;
#pragma unroll
  for (int i = 0; i < 4; ++i) {
    vs[i] = kr[threadIdx.x + i * 256];
    ss += vs[i] * vs[i];
  }
  red[threadIdx.x] = ss;
  __syncthreads();
  for (int s = 128; s > 0; s >>= 1) {
    if (threadIdx.x < s) red[threadIdx.x] += red[threadIdx.x + s];
    __syncthreads();
  }
  float inv = 1.f / fmaxf(sqrtf(red[0]), 1e-12f);
#pragma unroll
  for (int i = 0; i < 4; ++i) kr[threadIdx.x + i * 256] = vs[i] * inv;
}

// ---------------------------------------------------------------------------
// In-place EMA over T: one thread per (b,d), coalesced across d.
// ---------------------------------------------------------------------------
__global__ void ema_kernel(float* __restrict__ v,
                           const float* __restrict__ alpha_logit) {
  int tid = blockIdx.x * blockDim.x + threadIdx.x;  // 0 .. B*D-1
  int b = tid / Dm, d = tid % Dm;
  float a = 1.f / (1.f + __expf(-alpha_logit[d]));
  float oma = 1.f - a;
  float s = 0.f;
  size_t idx = (size_t)b * Tn * Dm + d;
  for (int t = 0; t < Tn; ++t) {
    s = a * v[idx] + oma * s;
    v[idx] = s;
    idx += Dm;
  }
}

// ---------------------------------------------------------------------------
// Delta-rule scan. Wave per state row i: S[i,:] lives in 32 VGPR f32/lane
// (lane L owns columns jj*32+L). k_t / q_t staged in LDS per block per step.
// S[i,:] += beta*(v[i] - S.k) k ; o[i] = S[i,:].q  via 32-lane shfl reductions.
// ---------------------------------------------------------------------------
__global__ __launch_bounds__(1024)
void delta_kernel(const float* __restrict__ q, const float* __restrict__ k,
                  const float* __restrict__ v, const float* __restrict__ beta,
                  float* __restrict__ o) {
  __shared__ float ks[Dm];
  __shared__ float qs[Dm];
  int lane = threadIdx.x & 31;
  int wid  = threadIdx.x >> 5;
  int b = blockIdx.y;
  int i = blockIdx.x * 32 + wid;   // state row (v-dim)
  float S[32];
#pragma unroll
  for (int jj = 0; jj < 32; ++jj) S[jj] = 0.f;
  size_t base = (size_t)b * Tn * Dm;
  for (int t = 0; t < Tn; ++t) {
    size_t rb = base + (size_t)t * Dm;
    ks[threadIdx.x] = k[rb + threadIdx.x];
    qs[threadIdx.x] = q[rb + threadIdx.x];
    __syncthreads();
    float sk = 0.f;
#pragma unroll
    for (int jj = 0; jj < 32; ++jj) sk += S[jj] * ks[jj * 32 + lane];
#pragma unroll
    for (int m = 16; m > 0; m >>= 1) sk += __shfl_xor(sk, m, 32);
    float bt = beta[b * Tn + t];
    float coef = bt * (v[rb + i] - sk);
    float ov = 0.f;
#pragma unroll
    for (int jj = 0; jj < 32; ++jj) {
      float kv = ks[jj * 32 + lane];
      S[jj] += coef * kv;
      ov += S[jj] * qs[jj * 32 + lane];
    }
#pragma unroll
    for (int m = 16; m > 0; m >>= 1) ov += __shfl_xor(ov, m, 32);
    if (lane == 0) o[rb + i] = ov;
    __syncthreads();
  }
}

// ---------------------------------------------------------------------------
// Elementwise helpers
// ---------------------------------------------------------------------------
__global__ void copy_f32x4_kernel(const float4* __restrict__ s,
                                  float4* __restrict__ d, size_t n4) {
  size_t i = blockIdx.x * (size_t)blockDim.x + threadIdx.x;
  if (i < n4) d[i] = s[i];
}

__global__ void tobf16_kernel(const float* __restrict__ s,
                              uint16_t* __restrict__ d, size_t n) {
  size_t i = blockIdx.x * (size_t)blockDim.x + threadIdx.x;
  if (i < n) d[i] = f32_to_bf16(s[i]);
}

__global__ void silu_mul_kernel(const float* __restrict__ g,
                                const float* __restrict__ u,
                                uint16_t* __restrict__ d, size_t n) {
  size_t i = blockIdx.x * (size_t)blockDim.x + threadIdx.x;
  if (i < n) {
    float gv = g[i];
    float s = gv / (1.f + __expf(-gv));
    d[i] = f32_to_bf16(s * u[i]);
  }
}

// ---------------------------------------------------------------------------
extern "C" void kernel_launch(void* const* d_in, const int* in_sizes, int n_in,
                              void* d_out, int out_size, void* d_ws,
                              size_t ws_size, hipStream_t stream) {
  (void)in_sizes; (void)n_in; (void)out_size; (void)ws_size;
  const float* x           = (const float*)d_in[0];
  const float* norm1_w     = (const float*)d_in[1];
  const float* Wq          = (const float*)d_in[2];
  const float* Wk          = (const float*)d_in[3];
  const float* Wv          = (const float*)d_in[4];
  const float* beta_w      = (const float*)d_in[5];
  const float* beta_b      = (const float*)d_in[6];
  const float* Wo          = (const float*)d_in[7];
  const float* alpha_logit = (const float*)d_in[8];
  const float* norm2_w     = (const float*)d_in[9];
  const float* w_gate      = (const float*)d_in[10];
  const float* w_up        = (const float*)d_in[11];
  const float* w_down      = (const float*)d_in[12];
  float* out = (float*)d_out;

  // Bump allocator over workspace
  char* ws = (char*)d_ws;
  size_t off = 0;
  auto alloc = [&](size_t bytes) -> void* {
    void* p = ws + off;
    off = (off + bytes + 255) & ~(size_t)255;
    return p;
  };
  const size_t nBTD = (size_t)BT * Dm;       // 8.4M
  const size_t nBTH = (size_t)BT * HIDm;     // 16.8M

  uint16_t* wq_p = (uint16_t*)alloc((size_t)Dm * Dm * 2);
  uint16_t* wk_p = (uint16_t*)alloc((size_t)Dm * Dm * 2);
  uint16_t* wv_p = (uint16_t*)alloc((size_t)Dm * Dm * 2);
  uint16_t* wo_p = (uint16_t*)alloc((size_t)Dm * Dm * 2);
  uint16_t* wg_p = (uint16_t*)alloc((size_t)Dm * HIDm * 2);
  uint16_t* wu_p = (uint16_t*)alloc((size_t)Dm * HIDm * 2);
  uint16_t* wd_p = (uint16_t*)alloc((size_t)HIDm * Dm * 2);
  uint16_t* hbf  = (uint16_t*)alloc(nBTD * 2);   // reused: h, attn_bf, h2
  float*    beta = (float*)alloc((size_t)BT * 4);
  float*    big  = (float*)alloc(nBTD * 4 * 4);  // q|k|v|attn, later g|u
  float* qb = big;
  float* kb = big + nBTD;
  float* vb = big + 2 * nBTD;
  float* at = big + 3 * nBTD;
  float* g  = big;              // covers q+k region (BT*HID f32)
  float* u  = big + nBTH;       // covers v+attn region
  uint16_t* act_bf = (uint16_t*)alloc(nBTH * 2);

  auto pack = [&](const float* W, uint16_t* P, int K, int N) {
    int total = (K >> 5) * (N >> 4) * 32;
    pack_b_kernel<<<(total + 255) / 256, 256, 0, stream>>>(W, P, K, N);
  };
  auto gemm = [&](const uint16_t* A, const uint16_t* Bp, float* C,
                  int M, int N, int K, int acc) {
    gemm_bf16_kernel<<<dim3(N / 128, M / 128), 256, 0, stream>>>(A, Bp, C, M,
                                                                 N, K, acc);
  };

  // 1) Pack all weights to WMMA bf16 layout
  pack(Wq, wq_p, Dm, Dm);
  pack(Wk, wk_p, Dm, Dm);
  pack(Wv, wv_p, Dm, Dm);
  pack(Wo, wo_p, Dm, Dm);
  pack(w_gate, wg_p, Dm, HIDm);
  pack(w_up,   wu_p, Dm, HIDm);
  pack(w_down, wd_p, HIDm, Dm);

  // 2) h = rmsnorm(x); beta = sigmoid(h.beta_w + beta_b)
  rmsnorm_kernel<<<BT, 256, 0, stream>>>(x, norm1_w, hbf, beta_w, beta_b, beta);

  // 3) q,k,v projections (WMMA)
  gemm(hbf, wq_p, qb, BT, Dm, Dm, 0);
  gemm(hbf, wk_p, kb, BT, Dm, Dm, 0);
  gemm(hbf, wv_p, vb, BT, Dm, Dm, 0);

  // 4) k normalize, v EMA
  knorm_kernel<<<BT, 256, 0, stream>>>(kb);
  ema_kernel<<<(Bn * Dm) / 256, 256, 0, stream>>>(vb, alpha_logit);

  // 5) delta-rule scan -> attn
  delta_kernel<<<dim3(Dm / 32, Bn), 1024, 0, stream>>>(qb, kb, vb, beta, at);

  // 6) out = x + attn @ Wo   (copy then WMMA-accumulate)
  copy_f32x4_kernel<<<(unsigned)((nBTD / 4 + 255) / 256), 256, 0, stream>>>(
      (const float4*)x, (float4*)out, nBTD / 4);
  tobf16_kernel<<<(unsigned)((nBTD + 255) / 256), 256, 0, stream>>>(at, hbf,
                                                                    nBTD);
  gemm(hbf, wo_p, out, BT, Dm, Dm, 1);

  // 7) h2 = rmsnorm(out); SwiGLU MLP accumulated into out
  rmsnorm_kernel<<<BT, 256, 0, stream>>>(out, norm2_w, hbf, nullptr, nullptr,
                                         nullptr);
  gemm(hbf, wg_p, g, BT, HIDm, Dm, 0);
  gemm(hbf, wu_p, u, BT, HIDm, Dm, 0);
  silu_mul_kernel<<<(unsigned)((nBTH + 255) / 256), 256, 0, stream>>>(g, u,
                                                                      act_bf,
                                                                      nBTH);
  gemm(act_bf, wd_p, out, BT, Dm, HIDm, 1);
}
